// NodeEmbedding_59622736003308
// MI455X (gfx1250) — matrix operand order repaired and checked by reading, not compiled, not run on hardware
//
#include <hip/hip_runtime.h>
#include <hip/hip_bf16.h>

typedef __attribute__((ext_vector_type(16))) _Float16 v16h;
typedef __attribute__((ext_vector_type(4)))  _Float16 v4h;
typedef __attribute__((ext_vector_type(8)))  float    v8f;

constexpr int HID   = 128;
constexpr int RBF   = 64;
constexpr int MAXZ1 = 101;
constexpr float CUT = 5.0f;
constexpr float PIF = 3.14159265358979323846f;

// ---- gfx1250 async global->LDS copy (ASYNCcnt-tracked DMA path) ----
__device__ __forceinline__ unsigned lds_off(const void* p) {
    // low 32 bits of an LDS generic address == wave-relative LDS byte offset
    return (unsigned)(unsigned long long)p;
}
__device__ __forceinline__ void async_g2l_b128(unsigned ldsoff, const void* g) {
    asm volatile("global_load_async_to_lds_b128 %0, %1, off"
                 :: "v"(ldsoff), "v"(g) : "memory");
}
__device__ __forceinline__ void wait_async0() {
#if __has_builtin(__builtin_amdgcn_s_wait_asynccnt)
    __builtin_amdgcn_s_wait_asynccnt(0);
#else
    asm volatile("s_wait_asynccnt 0x0" ::: "memory");
#endif
}

// ---------------- zero workspace (agg accumulator) ----------------
__global__ void zero_kernel(float* __restrict__ p, int n) {
    int i = blockIdx.x * blockDim.x + threadIdx.x;
    if (i < n) p[i] = 0.0f;
}

// ---------------- edge kernel ----------------
// W = (attr @ projW.T + b) * C(dist);  msg = W * nbr[z[col]];  agg[row] += msg
__global__ void edge_msg_kernel(const float* __restrict__ edge_attr,
                                const float* __restrict__ edge_dist,
                                const int*   __restrict__ edge_index,  // (2,E) flat
                                const int*   __restrict__ z,
                                const float* __restrict__ neighbor_emb,
                                const float* __restrict__ proj_W,      // (HID,RBF) row-major == B^T K-contig
                                const float* __restrict__ proj_b,
                                float* __restrict__ agg,               // (N,HID)
                                int E, int nTiles)
{
    extern __shared__ unsigned char smem[];
    float*    nbrTab = (float*)smem;                         // 51712 B
    _Float16* Bt     = (_Float16*)(smem + 51712);            // 16384 B
    float*    bias   = (float*)(smem + 68096);               // 512 B
    float*    Afz    = (float*)(smem + 68608);               // 16x64 f32 = 4096 B
    float*    C16    = (float*)(smem + 72704);               // 64 B
    int*      nzo    = (int*)  (smem + 72768);               // 64 B (z[col]*HID)
    int*      rowOff = (int*)  (smem + 72832);               // 64 B (row*HID)

    const int tid = threadIdx.x;
    // stage constant tables (vectorized)
    for (int i = tid * 4; i < MAXZ1 * HID; i += blockDim.x * 4)
        *(float4*)&nbrTab[i] = *(const float4*)&neighbor_emb[i];
    for (int i = tid * 4; i < HID * RBF; i += blockDim.x * 4) {
        float4 w = *(const float4*)&proj_W[i];
        v4h h; h[0] = (_Float16)w.x; h[1] = (_Float16)w.y;
               h[2] = (_Float16)w.z; h[3] = (_Float16)w.w;
        *(v4h*)&Bt[i] = h;
    }
    for (int i = tid; i < HID; i += blockDim.x) bias[i] = proj_b[i];
    __syncthreads();

    const int lane    = tid & 31;
    const int wave    = tid >> 5;          // 0..7 -> 16-wide hidden tile
    const int halfSel = lane >> 4;
    const int mlane   = lane & 15;
    const int ncol    = wave * 16 + mlane;
    const int mbase   = halfSel << 3;
    const float biasv = bias[ncol];
    const unsigned afzOff = lds_off(Afz);

    for (int t = blockIdx.x; t < nTiles; t += gridDim.x) {
        const int e0 = t * 16;
        if (tid < 16) {
            int e = e0 + tid;
            if (e < E) {
                float d = edge_dist[e];
                float c = 0.5f * (cosf(PIF * d / CUT) + 1.0f);
                C16[tid]    = (d < CUT) ? c : 0.0f;   // tail rows get C=0 -> val=0
                rowOff[tid] = edge_index[e] * HID;
                nzo[tid]    = z[edge_index[E + e]] * HID;
            } else { C16[tid] = 0.0f; rowOff[tid] = 0; nzo[tid] = 0; }
        }
        if (e0 + 16 <= E) {
            // async DMA: 256 threads x 16B = whole 16x64 f32 tile
            async_g2l_b128(afzOff + tid * 16, edge_attr + e0 * RBF + tid * 4);
        } else {
            for (int i = tid; i < 16 * RBF; i += blockDim.x) {
                int e = e0 + (i >> 6);
                Afz[i] = (e < E) ? edge_attr[e * RBF + (i & 63)] : 0.0f;
            }
        }
        wait_async0();
        __syncthreads();

        v8f acc = {};
#pragma unroll
        for (int k0 = 0; k0 < RBF; k0 += 32) {
            v16h a, b;
            const int ab = mlane * RBF + k0 + (halfSel << 4);
#pragma unroll
            for (int i = 0; i < 8; ++i) {
                a[i]     = (_Float16)Afz[ab + i];
                a[8 + i] = (_Float16)Afz[ab + 8 + i];
            }
            const int bb = ncol * RBF + k0 + (halfSel << 4);
#pragma unroll
            for (int i = 0; i < 16; ++i) b[i] = Bt[bb + i];
            acc = __builtin_amdgcn_wmma_f32_16x16x32_f16(false, a, false, b,
                                                         (short)0, acc, false, false);
        }
        // branchless epilogue: tail rows add exactly 0.0f to agg[0..]
#pragma unroll
        for (int v = 0; v < 8; ++v) {
            int m = mbase + v;
            float val = (acc[v] + biasv) * C16[m] * nbrTab[nzo[m] + ncol];
            __hip_atomic_fetch_add(&agg[rowOff[m] + ncol], val,
                                   __ATOMIC_RELAXED, __HIP_MEMORY_SCOPE_AGENT);
        }
        __syncthreads();
    }
}

// ---------------- combine kernel: out = [x | agg] @ comb_W.T + b ----------------
__global__ void combine_kernel(const int*   __restrict__ z,
                               const float* __restrict__ atom_emb,    // (101,HID)
                               const float* __restrict__ agg,         // (N,HID)
                               const float* __restrict__ comb_W,      // (HID,2HID) row-major == B^T K-contig
                               const float* __restrict__ comb_b,
                               float* __restrict__ out,               // (N,HID)
                               int N, int nTiles)
{
    extern __shared__ unsigned char smem[];
    _Float16* Bt   = (_Float16*)smem;                  // 128*256 f16 = 65536 B
    float*    bias = (float*)(smem + 65536);           // 512 B
    float*    Axz  = (float*)(smem + 66048);           // 16x128 f32 = 8192 B
    float*    Agz  = (float*)(smem + 74240);           // 16x128 f32 = 8192 B

    const int tid = threadIdx.x;
    for (int i = tid * 4; i < HID * 2 * HID; i += blockDim.x * 4) {
        float4 w = *(const float4*)&comb_W[i];
        v4h h; h[0] = (_Float16)w.x; h[1] = (_Float16)w.y;
               h[2] = (_Float16)w.z; h[3] = (_Float16)w.w;
        *(v4h*)&Bt[i] = h;
    }
    for (int i = tid; i < HID; i += blockDim.x) bias[i] = comb_b[i];
    __syncthreads();

    const int lane    = tid & 31;
    const int wave    = tid >> 5;
    const int halfSel = lane >> 4;
    const int mlane   = lane & 15;
    const int ncol    = wave * 16 + mlane;
    const int mbase   = halfSel << 3;
    const float biasv = bias[ncol];
    const unsigned axzOff = lds_off(Axz);
    const unsigned agzOff = lds_off(Agz);

    for (int t = blockIdx.x; t < nTiles; t += gridDim.x) {
        const int n0 = t * 16;
        const bool full = (n0 + 16 <= N);
        if (full) {
            // Agz: 16 contiguous rows of agg -> 8KB, async DMA (2 x b128 per thread)
            // Axz: gathered atom_emb rows, per-16B-chunk async DMA
#pragma unroll
            for (int j = 0; j < 2; ++j) {
                int c = tid + j * 256;               // chunk id, 512 chunks of 16B
                async_g2l_b128(agzOff + c * 16, agg + n0 * HID + c * 4);
                int m = c >> 5, k4 = (c & 31) * 4;
                async_g2l_b128(axzOff + c * 16, atom_emb + z[n0 + m] * HID + k4);
            }
        } else {
            for (int i = tid; i < 16 * HID; i += blockDim.x) {
                int m = i >> 7, k = i & 127, node = n0 + m;
                if (node < N) {
                    Axz[i] = atom_emb[z[node] * HID + k];
                    Agz[i] = agg[node * HID + k];
                } else { Axz[i] = 0.0f; Agz[i] = 0.0f; }
            }
        }
        wait_async0();
        __syncthreads();

        v8f acc = {};
#pragma unroll
        for (int k0 = 0; k0 < HID; k0 += 32) {
            v16h a, b;
            const int ab = mlane * HID + k0 + (halfSel << 4);
            const int bb = ncol * 2 * HID + k0 + (halfSel << 4);
            // x panel (K = 0..127)
#pragma unroll
            for (int i = 0; i < 8; ++i) {
                a[i]     = (_Float16)Axz[ab + i];
                a[8 + i] = (_Float16)Axz[ab + 8 + i];
            }
#pragma unroll
            for (int i = 0; i < 16; ++i) b[i] = Bt[bb + i];
            acc = __builtin_amdgcn_wmma_f32_16x16x32_f16(false, a, false, b,
                                                         (short)0, acc, false, false);
            // agg panel (K = 128..255)
#pragma unroll
            for (int i = 0; i < 8; ++i) {
                a[i]     = (_Float16)Agz[ab + i];
                a[8 + i] = (_Float16)Agz[ab + 8 + i];
            }
#pragma unroll
            for (int i = 0; i < 16; ++i) b[i] = Bt[bb + HID + i];
            acc = __builtin_amdgcn_wmma_f32_16x16x32_f16(false, a, false, b,
                                                         (short)0, acc, false, false);
        }
        if (full) {
#pragma unroll
            for (int v = 0; v < 8; ++v)
                out[(n0 + mbase + v) * HID + ncol] = acc[v] + biasv;
        } else {
#pragma unroll
            for (int v = 0; v < 8; ++v) {
                int node = n0 + mbase + v;
                if (node < N) out[node * HID + ncol] = acc[v] + biasv;
            }
        }
        __syncthreads();
    }
}

extern "C" void kernel_launch(void* const* d_in, const int* in_sizes, int n_in,
                              void* d_out, int out_size, void* d_ws, size_t ws_size,
                              hipStream_t stream) {
    const int*   z     = (const int*)d_in[0];
    const int*   eidx  = (const int*)d_in[1];
    const float* edist = (const float*)d_in[2];
    const float* eattr = (const float*)d_in[3];
    const float* aemb  = (const float*)d_in[4];
    const float* nemb  = (const float*)d_in[5];
    const float* pW    = (const float*)d_in[6];
    const float* pb    = (const float*)d_in[7];
    const float* cW    = (const float*)d_in[8];
    const float* cb    = (const float*)d_in[9];
    float*       out   = (float*)d_out;
    float*       agg   = (float*)d_ws;

    const int N = in_sizes[0];
    const int E = in_sizes[2];

    // 1) zero the L2-resident accumulator (required each call: atomics + graph replay)
    int aggN = N * HID;
    zero_kernel<<<(aggN + 255) / 256, 256, 0, stream>>>(agg, aggN);

    // 2) edge GEMM + gather-multiply + scatter-add
    int eTiles  = (E + 15) / 16;
    int eBlocks = eTiles < 2048 ? eTiles : 2048;
    edge_msg_kernel<<<eBlocks, 256, 72896, stream>>>(eattr, edist, eidx, z, nemb,
                                                     pW, pb, agg, E, eTiles);

    // 3) node GEMM: [x | agg] @ comb_W.T + b
    int nTiles  = (N + 15) / 16;
    int cBlocks = nTiles < 1024 ? nTiles : 1024;
    combine_kernel<<<cBlocks, 256, 82432, stream>>>(z, aemb, agg, cW, cb, out, N, nTiles);
}
